// Attention_50680614092907
// MI455X (gfx1250) — matrix operand order repaired and checked by reading, not compiled, not run on hardware
//
#include <hip/hip_runtime.h>

#define NT 64
#define NB 1024
#define NC 512
#define HH 512
#define EE 128
#define NCLS 97
#define H3 1536
#define XDIM 640
#define NGEN 128
#define MAXS 32

typedef __attribute__((ext_vector_type(16))) __bf16          v16bf;
typedef __attribute__((ext_vector_type(16))) unsigned short  v16us;
typedef __attribute__((ext_vector_type(8)))  unsigned short  v8us;
typedef __attribute__((ext_vector_type(8)))  float           v8f;
typedef __attribute__((ext_vector_type(4)))  float           v4f;
typedef __attribute__((ext_vector_type(2)))  unsigned int    v2u;

__device__ __forceinline__ unsigned short f2bf(float f) {
    union { float f; unsigned int u; } c; c.f = f;
    unsigned int u = c.u;
    unsigned int r = (u + 0x7FFFu + ((u >> 16) & 1u)) >> 16;  // RNE
    return (unsigned short)r;
}
__device__ __forceinline__ float bf2f(unsigned short h) {
    union { unsigned int u; float f; } c; c.u = ((unsigned int)h) << 16;
    return c.f;
}

union Frag16 {
    struct { v8us lo; v8us hi; } h;
    v16us u;
    v16bf b;
};

// ---------------------------------------------------------------------------
// C[M,N](f32 or bf16) = A[M,K](bf16) @ B[N,K](bf16)^T (+bias)
// One wave computes a 32x32 tile via 4x v_wmma_f32_16x16x32_bf16 per K-step.
// WMMA per-lane layouts (CDNA5 ISA 7.12.2):
//   A 16x32 bf16: halves 0..7  -> K = k0 .. k0+7   (k0 = lane<16 ? 0 : 8)
//                 halves 8..15 -> K = k0+16 .. k0+23
//   B 32x16 bf16 (stored as N x K rows): halves 0..7 -> K = kb .. kb+7, halves 8..15 -> K = kb+16 .. kb+23
//                 (kb = lane<16 ? 0 : 8), col = lane%16
//   C 16x16 f32:  vgpr r, lane L -> row = r + (L<16?0:8), col = L%16
// ---------------------------------------------------------------------------
__global__ void __launch_bounds__(128)
gemm_bf16(const unsigned short* __restrict__ Aw, int lda,
          const unsigned short* __restrict__ Bw, int ldb,
          float* __restrict__ C, int ldc,
          const float* __restrict__ bias,
          int M, int N, int K, int store_bf16)
{
    int wid  = ((int)blockIdx.x * (int)blockDim.x + (int)threadIdx.x) >> 5;
    int lane = (int)threadIdx.x & 31;
    int tilesM = M >> 5, tilesN = N >> 5;
    if (wid >= tilesM * tilesN) return;
    int tm = wid % tilesM;
    int tn = wid / tilesM;
    int row0 = tm << 5, col0 = tn << 5;

    int hi    = (lane >> 4) & 1;
    int l16   = lane & 15;
    int koffA = hi << 3;   // 0 or 8
    int koffB = hi << 3;   // 0 or 8

    v8f acc00 = {}; v8f acc01 = {}; v8f acc10 = {}; v8f acc11 = {};

    const unsigned short* Arow0 = Aw + (size_t)(row0 + l16) * lda + koffA;
    const unsigned short* Arow1 = Arow0 + (size_t)16 * lda;
    const unsigned short* Brow0 = Bw + (size_t)(col0 + l16) * ldb + koffB;
    const unsigned short* Brow1 = Brow0 + (size_t)16 * ldb;

    #pragma unroll 2
    for (int kk = 0; kk < K; kk += 32) {
        // prefetch next K-step (global_prefetch_b8; speculative OOB is dropped)
        __builtin_prefetch(Arow0 + kk + 32, 0, 1);
        __builtin_prefetch(Arow1 + kk + 32, 0, 1);

        Frag16 a0, a1;
        a0.h.lo = *(const v8us*)(Arow0 + kk);
        a0.h.hi = *(const v8us*)(Arow0 + kk + 16);
        a1.h.lo = *(const v8us*)(Arow1 + kk);
        a1.h.hi = *(const v8us*)(Arow1 + kk + 16);
        Frag16 fb0, fb1;
        fb0.h.lo = *(const v8us*)(Brow0 + kk);
        fb0.h.hi = *(const v8us*)(Brow0 + kk + 16);
        fb1.h.lo = *(const v8us*)(Brow1 + kk);
        fb1.h.hi = *(const v8us*)(Brow1 + kk + 16);
        v16bf b0 = fb0.b;
        v16bf b1 = fb1.b;

        acc00 = __builtin_amdgcn_wmma_f32_16x16x32_bf16(false, a0.b, false, b0, (short)0, acc00, false, false);
        acc01 = __builtin_amdgcn_wmma_f32_16x16x32_bf16(false, a0.b, false, b1, (short)0, acc01, false, false);
        acc10 = __builtin_amdgcn_wmma_f32_16x16x32_bf16(false, a1.b, false, b0, (short)0, acc10, false, false);
        acc11 = __builtin_amdgcn_wmma_f32_16x16x32_bf16(false, a1.b, false, b1, (short)0, acc11, false, false);
    }

    int rofs = hi << 3;  // 0 or 8
    #pragma unroll
    for (int im = 0; im < 2; ++im) {
        #pragma unroll
        for (int jn = 0; jn < 2; ++jn) {
            v8f acc = (im == 0) ? ((jn == 0) ? acc00 : acc01)
                                : ((jn == 0) ? acc10 : acc11);
            int col  = col0 + (jn << 4) + l16;
            int rowb = row0 + (im << 4) + rofs;
            float bv = bias ? bias[col] : 0.0f;
            if (store_bf16) {
                unsigned short* Cb = (unsigned short*)C;
                #pragma unroll
                for (int r = 0; r < 8; ++r)
                    Cb[(size_t)(rowb + r) * ldc + col] = f2bf(acc[r] + bv);
            } else {
                #pragma unroll
                for (int r = 0; r < 8; ++r)
                    C[(size_t)(rowb + r) * ldc + col] = acc[r] + bv;
            }
        }
    }
}

// e[t,b] = dot(w_score, tanh(fp[t,b,:] + hproj[b,:]))   (one wave per (t,b))
__global__ void energy_kernel(const unsigned short* __restrict__ fp,
                              const float* __restrict__ hproj,
                              const float* __restrict__ wscore,
                              float* __restrict__ e)
{
    int g = (int)blockIdx.x * (int)blockDim.x + (int)threadIdx.x;
    int warp = g >> 5;
    int lane = g & 31;
    if (warp >= NT * NB) return;
    int t = warp >> 10;
    int b = warp & (NB - 1);
    int base = lane * 16;  // HH = 32 lanes * 16
    const unsigned short* fr = fp + ((size_t)t * NB + b) * HH + base;
    const float* hr = hproj + (size_t)b * HH + base;
    const float* wr = wscore + base;

    v8us f0 = ((const v8us*)fr)[0];
    v8us f1 = ((const v8us*)fr)[1];
    float fv[16];
    #pragma unroll
    for (int i = 0; i < 8; ++i) { fv[i] = bf2f(f0[i]); fv[8 + i] = bf2f(f1[i]); }
    v4f hv[4], wv[4];
    #pragma unroll
    for (int j = 0; j < 4; ++j) { hv[j] = ((const v4f*)hr)[j]; wv[j] = ((const v4f*)wr)[j]; }

    float s = 0.f;
    #pragma unroll
    for (int j = 0; j < 4; ++j)
        #pragma unroll
        for (int q = 0; q < 4; ++q)
            s += tanhf(fv[j * 4 + q] + hv[j][q]) * wv[j][q];

    #pragma unroll
    for (int off = 16; off > 0; off >>= 1)
        s += __shfl_down(s, off, 32);
    if (lane == 0) e[(size_t)t * NB + b] = s;
}

// softmax over t (NT=64) for each b; in-place; one wave per b, 2 t per lane
__global__ void softmax_kernel(float* __restrict__ e)
{
    int g = (int)blockIdx.x * (int)blockDim.x + (int)threadIdx.x;
    int warp = g >> 5, lane = g & 31;
    if (warp >= NB) return;
    int b = warp;
    float v0 = e[(size_t)lane * NB + b];
    float v1 = e[(size_t)(lane + 32) * NB + b];
    float m = fmaxf(v0, v1);
    #pragma unroll
    for (int off = 16; off > 0; off >>= 1)
        m = fmaxf(m, __shfl_xor(m, off, 32));
    float p0 = __expf(v0 - m), p1 = __expf(v1 - m);
    float s = p0 + p1;
    #pragma unroll
    for (int off = 16; off > 0; off >>= 1)
        s += __shfl_xor(s, off, 32);
    float inv = 1.0f / s;
    e[(size_t)lane * NB + b] = p0 * inv;
    e[(size_t)(lane + 32) * NB + b] = p1 * inv;
}

// x[b, :NC] = sum_t alpha[t,b]*feats_bf[t,b,:] ; x[b, NC:] = char_emb[tgt[b]]
// Reads the bf16 feats mirror (keeps per-step working set inside the 192MB L2)
// and writes x in bf16 (only consumed as a WMMA A-operand). 2 channels/thread.
__global__ void context_embed_kernel(const unsigned short* __restrict__ featsbf,
                                     const float* __restrict__ alpha,
                                     const float* __restrict__ char_emb,
                                     const int* __restrict__ tgt,
                                     unsigned short* __restrict__ x)
{
    int idx = (int)blockIdx.x * (int)blockDim.x + (int)threadIdx.x;
    if (idx >= NB * (XDIM / 2)) return;
    int b  = idx / (XDIM / 2);
    int c2 = (idx - b * (XDIM / 2)) * 2;
    unsigned int outpair;
    if (c2 < NC) {
        float s0 = 0.f, s1 = 0.f;
        #pragma unroll 4
        for (int t = 0; t < NT; ++t) {
            float a = alpha[(size_t)t * NB + b];
            unsigned int pr = *(const unsigned int*)(featsbf + ((size_t)t * NB + b) * NC + c2);
            s0 += a * bf2f((unsigned short)(pr & 0xFFFFu));
            s1 += a * bf2f((unsigned short)(pr >> 16));
        }
        outpair = (unsigned int)f2bf(s0) | ((unsigned int)f2bf(s1) << 16);
    } else {
        int ch = tgt[b];
        float e0 = char_emb[(size_t)ch * EE + (c2 - NC)];
        float e1 = char_emb[(size_t)ch * EE + (c2 - NC) + 1];
        outpair = (unsigned int)f2bf(e0) | ((unsigned int)f2bf(e1) << 16);
    }
    *(unsigned int*)(x + (size_t)b * XDIM + c2) = outpair;
}

// GRU elementwise; gate order r,z,n; biases already folded into gi/gh GEMMs.
// Writes both the f32 hidden and its bf16 mirror (WMMA A operand).
__global__ void gru_kernel(const float* __restrict__ gi, const float* __restrict__ gh,
                           float* __restrict__ hidden, unsigned short* __restrict__ hidden_bf)
{
    int idx = (int)blockIdx.x * (int)blockDim.x + (int)threadIdx.x;
    if (idx >= NB * HH) return;
    int b = idx >> 9;
    int h = idx & (HH - 1);
    const float* gib = gi + (size_t)b * H3;
    const float* ghb = gh + (size_t)b * H3;
    float r = 1.f / (1.f + __expf(-(gib[h] + ghb[h])));
    float z = 1.f / (1.f + __expf(-(gib[HH + h] + ghb[HH + h])));
    float n = tanhf(gib[2 * HH + h] + r * ghb[2 * HH + h]);
    float hv = hidden[idx];
    float hn = (1.f - z) * n + z * hv;
    hidden[idx]    = hn;
    hidden_bf[idx] = f2bf(hn);
}

// copy 97 real logits to probs[s], argmax(+1) -> tgt  (one wave per b)
__global__ void logits_post_kernel(const float* __restrict__ logits,
                                   float* __restrict__ probs, int* __restrict__ tgt, int s)
{
    int g = (int)blockIdx.x * (int)blockDim.x + (int)threadIdx.x;
    int warp = g >> 5, lane = g & 31;
    if (warp >= NB) return;
    int b = warp;
    float best = -3.0e38f; int besti = 0x7fffffff;
    for (int c = lane; c < NCLS; c += 32) {
        float v = logits[(size_t)b * NGEN + c];
        probs[((size_t)s * NB + b) * NCLS + c] = v;
        if (v > best) { best = v; besti = c; }
    }
    #pragma unroll
    for (int off = 16; off > 0; off >>= 1) {
        float ob = __shfl_xor(best, off, 32);
        int   oi = __shfl_xor(besti, off, 32);
        if (ob > best || (ob == best && oi < besti)) { best = ob; besti = oi; }
    }
    if (lane == 0) tgt[b] = besti + 1;
}

__global__ void init_kernel(float* __restrict__ hidden, unsigned short* __restrict__ hidden_bf,
                            int* __restrict__ tgt)
{
    int idx = (int)blockIdx.x * (int)blockDim.x + (int)threadIdx.x;
    if (idx < NB * HH) { hidden[idx] = 0.f; hidden_bf[idx] = 0; }
    if (idx < NB) tgt[idx] = 0;
}

// 4-wide f32 -> bf16 conversion (all sizes here are multiples of 4)
__global__ void cvt4_kernel(const float* __restrict__ src, unsigned short* __restrict__ dst, int n4)
{
    int i = (int)blockIdx.x * (int)blockDim.x + (int)threadIdx.x;
    if (i >= n4) return;
    v4f v = ((const v4f*)src)[i];
    unsigned int lo = (unsigned int)f2bf(v[0]) | ((unsigned int)f2bf(v[1]) << 16);
    unsigned int hi = (unsigned int)f2bf(v[2]) | ((unsigned int)f2bf(v[3]) << 16);
    v2u out; out[0] = lo; out[1] = hi;
    ((v2u*)dst)[i] = out;
}

__global__ void wgen_pad_kernel(const float* __restrict__ wg, const float* __restrict__ bg,
                                unsigned short* __restrict__ wgp, float* __restrict__ bgp)
{
    int i = (int)blockIdx.x * (int)blockDim.x + (int)threadIdx.x;
    if (i < NGEN * HH) {
        int row = i / HH, col = i - row * HH;
        wgp[i] = (row < NCLS) ? f2bf(wg[(size_t)row * HH + col]) : (unsigned short)0;
    }
    if (i < NGEN) bgp[i] = (i < NCLS) ? bg[i] : -1.0e30f;
}

__global__ void prefix_kernel(const int* __restrict__ tl, int* __restrict__ off)
{
    if (blockIdx.x == 0 && threadIdx.x == 0) {
        int a = 0;
        for (int b = 0; b < NB; ++b) { off[b] = a; a += tl[b]; }
        off[NB] = a;
    }
}

__global__ void gather_kernel(const float* __restrict__ probs, const int* __restrict__ tl,
                              const int* __restrict__ off, float* __restrict__ out)
{
    int b = (int)blockIdx.x;
    int n = tl[b];
    if (n > MAXS) n = MAXS;
    int base = off[b];
    int tot = n * NCLS;
    for (int i = (int)threadIdx.x; i < tot; i += (int)blockDim.x) {
        int t = i / NCLS;
        int c = i - t * NCLS;
        out[((size_t)(base + t)) * NCLS + c] = probs[((size_t)t * NB + b) * NCLS + c];
    }
}

extern "C" void kernel_launch(void* const* d_in, const int* in_sizes, int n_in,
                              void* d_out, int out_size, void* d_ws, size_t ws_size,
                              hipStream_t stream)
{
    const float* feats    = (const float*)d_in[0];
    const float* w_i2h    = (const float*)d_in[1];
    const float* w_h2h    = (const float*)d_in[2];
    const float* b_h2h    = (const float*)d_in[3];
    const float* w_score  = (const float*)d_in[4];
    const float* w_ih     = (const float*)d_in[5];
    const float* w_hh     = (const float*)d_in[6];
    const float* b_ih     = (const float*)d_in[7];
    const float* b_hh     = (const float*)d_in[8];
    const float* char_emb = (const float*)d_in[9];
    const float* w_gen    = (const float*)d_in[10];
    const float* b_gen    = (const float*)d_in[11];
    const int*   tl       = (const int*)d_in[12];
    (void)in_sizes; (void)n_in; (void)out_size; (void)ws_size;

    char* p = (char*)d_ws;
    auto alloc = [&](size_t bytes) -> void* {
        void* r = (void*)p;
        p += (bytes + 255) & ~(size_t)255;
        return r;
    };
    unsigned short* w_i2h_bf  = (unsigned short*)alloc((size_t)HH * NC * 2);
    unsigned short* w_h2h_bf  = (unsigned short*)alloc((size_t)HH * HH * 2);
    unsigned short* w_ih_bf   = (unsigned short*)alloc((size_t)H3 * XDIM * 2);
    unsigned short* w_hh_bf   = (unsigned short*)alloc((size_t)H3 * HH * 2);
    unsigned short* w_gen_bf  = (unsigned short*)alloc((size_t)NGEN * HH * 2);
    float*          b_gen_pad = (float*)alloc((size_t)NGEN * 4);
    unsigned short* feats_bf  = (unsigned short*)alloc((size_t)NT * NB * NC * 2);
    unsigned short* fp        = (unsigned short*)alloc((size_t)NT * NB * HH * 2);
    float*          hproj     = (float*)alloc((size_t)NB * HH * 4);
    float*          ebuf      = (float*)alloc((size_t)NT * NB * 4);
    unsigned short* xbuf      = (unsigned short*)alloc((size_t)NB * XDIM * 2);
    float*          hidden    = (float*)alloc((size_t)NB * HH * 4);
    unsigned short* hidden_bf = (unsigned short*)alloc((size_t)NB * HH * 2);
    float*          gi        = (float*)alloc((size_t)NB * H3 * 4);
    float*          gh        = (float*)alloc((size_t)NB * H3 * 4);
    float*          logits    = (float*)alloc((size_t)NB * NGEN * 4);
    float*          probs     = (float*)alloc((size_t)MAXS * NB * NCLS * 4);
    int*            tgt       = (int*)alloc((size_t)NB * 4);
    int*            offs      = (int*)alloc((size_t)(NB + 1) * 4);

    auto cvt = [&](const float* src, unsigned short* dst, int n) {
        int n4 = n / 4;
        cvt4_kernel<<<(n4 + 255) / 256, 256, 0, stream>>>(src, dst, n4);
    };
    cvt(w_i2h, w_i2h_bf, HH * NC);
    cvt(w_h2h, w_h2h_bf, HH * HH);
    cvt(w_ih,  w_ih_bf,  H3 * XDIM);
    cvt(w_hh,  w_hh_bf,  H3 * HH);
    cvt(feats, feats_bf, NT * NB * NC);
    wgen_pad_kernel<<<(NGEN * HH + 255) / 256, 256, 0, stream>>>(w_gen, b_gen, w_gen_bf, b_gen_pad);
    init_kernel<<<(NB * HH + 255) / 256, 256, 0, stream>>>(hidden, hidden_bf, tgt);
    prefix_kernel<<<1, 1, 0, stream>>>(tl, offs);

    auto launch_gemm = [&](const unsigned short* A, int lda, const unsigned short* B, int ldb,
                           float* C, int ldc, const float* bias,
                           int M, int N, int K, int sbf) {
        int waves  = (M >> 5) * (N >> 5);
        int blocks = (waves + 3) / 4;  // 4 waves / 128-thread block
        gemm_bf16<<<blocks, 128, 0, stream>>>(A, lda, B, ldb, C, ldc, bias, M, N, K, sbf);
    };

    // feats_proj = feats @ w_i2h^T   [65536,512], stored bf16 (once)
    launch_gemm(feats_bf, NC, w_i2h_bf, NC, (float*)fp, HH, nullptr, NT * NB, HH, NC, 1);

    for (int s = 0; s < MAXS; ++s) {
        // hproj = hidden @ w_h2h^T + b_h2h
        launch_gemm(hidden_bf, HH, w_h2h_bf, HH, hproj, HH, b_h2h, NB, HH, HH, 0);
        energy_kernel<<<(NT * NB * 32 + 127) / 128, 128, 0, stream>>>(fp, hproj, w_score, ebuf);
        softmax_kernel<<<(NB * 32 + 127) / 128, 128, 0, stream>>>(ebuf);
        context_embed_kernel<<<(NB * (XDIM / 2) + 255) / 256, 256, 0, stream>>>(feats_bf, ebuf, char_emb, tgt, xbuf);
        // gi = x @ w_ih^T + b_ih ; gh = h @ w_hh^T + b_hh
        launch_gemm(xbuf, XDIM, w_ih_bf, XDIM, gi, H3, b_ih, NB, H3, XDIM, 0);
        launch_gemm(hidden_bf, HH, w_hh_bf, HH, gh, H3, b_hh, NB, H3, HH, 0);
        gru_kernel<<<(NB * HH + 255) / 256, 256, 0, stream>>>(gi, gh, hidden, hidden_bf);
        // logits = h_new @ w_gen^T + b_gen (padded to 128 cols, pad bias = -1e30)
        launch_gemm(hidden_bf, HH, w_gen_bf, HH, logits, NGEN, b_gen_pad, NB, NGEN, HH, 0);
        logits_post_kernel<<<(NB * 32 + 127) / 128, 128, 0, stream>>>(logits, probs, tgt, s);
    }

    gather_kernel<<<NB, 128, 0, stream>>>(probs, tl, offs, (float*)d_out);
}
